// CrossDeformableAttentionBlockwFlow_14980845928998
// MI455X (gfx1250) — matrix operand order repaired and checked by reading, not compiled, not run on hardware
//
#include <hip/hip_runtime.h>
#include <hip/hip_bf16.h>
#include <math.h>

typedef __attribute__((ext_vector_type(16))) _Float16 v16h;
typedef __attribute__((ext_vector_type(8)))  float    v8f;

#define CONV_WAVES 8   // waves per conv workgroup, 16 Cout rows each

// ---------------------------------------------------------------------------
// Bilinear sample with zeros padding (per-corner validity, matches reference)
// ---------------------------------------------------------------------------
__device__ inline float bilin_zero(const float* __restrict__ p, int Hh, int Ww,
                                   float gx, float gy) {
    float x0f = floorf(gx), y0f = floorf(gy);
    float wx1 = gx - x0f, wy1 = gy - y0f;
    float acc = 0.f;
    #pragma unroll
    for (int dy = 0; dy < 2; ++dy) {
        #pragma unroll
        for (int dx = 0; dx < 2; ++dx) {
            float xf = x0f + (float)dx;
            float yf = y0f + (float)dy;
            float w = (dx ? wx1 : 1.f - wx1) * (dy ? wy1 : 1.f - wy1);
            bool valid = (xf >= 0.f) && (xf <= (float)(Ww - 1)) &&
                         (yf >= 0.f) && (yf <= (float)(Hh - 1));
            if (valid) {
                int xi = (int)xf, yi = (int)yf;
                acc += p[yi * Ww + xi] * w;
            }
        }
    }
    return acc;
}

// ---------------------------------------------------------------------------
// Implicit-GEMM conv via WMMA f16 -> f32, LDS-staged B tile.
//   Workgroup = 8 waves. All waves share one 16-pixel N tile; wave w owns
//   Cout rows [blockIdx.x*128 + w*16, +16). Per 32-wide K chunk the im2col
//   B tile (32K x 16N f16) is built cooperatively in LDS and consumed by all
//   waves (ds_load), weights stream as two contiguous 8-float spans per lane.
//   act: 0=linear 1=prelu(alpha) 2=2*tanh 3=gelu(tanh approx)
// ---------------------------------------------------------------------------
__global__ __launch_bounds__(32 * CONV_WAVES)
void conv_wmma_kernel(const float* __restrict__ in, const float* __restrict__ wgt,
                      const float* __restrict__ bias, const float* __restrict__ res,
                      const float* __restrict__ alpha, float* __restrict__ out,
                      int Bn, int Cin, int Cout, int Hh, int Ww, int Kk, int pad, int act)
{
    const int tid  = threadIdx.x;
    const int lane = tid & 31;
    const int wave = tid >> 5;
    const int g    = lane >> 4;     // lane-half
    const int hl   = lane & 15;
    const int HW   = Hh * Ww;
    const int KK   = Kk * Kk;
    const int Ktot = Cin * KK;

    const int tileM = (blockIdx.x * CONV_WAVES + wave) * 16;
    const int tileN = blockIdx.y * 16;

    // B-stage: [N][K] f16, row padded to 40 halfs for bank spread
    __shared__ _Float16 sB[16][40];

    // pixels this thread stages (n0 = tid>>5, n0+8) -- fixed across K chunks
    const int n0   = tid >> 5;          // 0..7
    const int kidx = tid & 31;          // K offset within chunk this thread fills
    int stN[2], stB[2], stY[2], stX[2];
    #pragma unroll
    for (int t = 0; t < 2; ++t) {
        int n = n0 + 8 * t;
        int gn = tileN + n;
        stN[t] = n;
        stB[t] = gn / HW;
        int rem = gn % HW;
        stY[t] = rem / Ww;
        stX[t] = rem % Ww;
    }

    // pixel owned by this lane for C stores (N = hl)
    const int n    = tileN + hl;
    const int bimg = n / HW;
    const int rem  = n % HW;

    const int coutA = tileM + hl;       // A-fragment row (M = hl)
    const float* rowA = wgt + (size_t)(coutA < Cout ? coutA : 0) * Ktot;

    v8f acc = {};

    for (int k0 = 0; k0 < Ktot; k0 += 32) {
        __syncthreads();   // previous chunk fully consumed
        // ---- cooperative im2col stage of B tile ----
        #pragma unroll
        for (int t = 0; t < 2; ++t) {
            int kk = k0 + kidx;
            float x = 0.f;
            if (kk < Ktot) {
                int cin = kk / KK;
                int r2  = kk % KK;
                int ky  = r2 / Kk, kx = r2 % Kk;
                int iy = stY[t] + ky - pad, ix = stX[t] + kx - pad;
                if (iy >= 0 && iy < Hh && ix >= 0 && ix < Ww)
                    x = in[(((size_t)(stB[t] * Cin + cin)) * Hh + iy) * Ww + ix];
            }
            sB[stN[t]][kidx] = (_Float16)x;
        }
        __syncthreads();

        // ---- A fragment: j=0..7 -> K=k0+8g+j ; j=8..15 -> K=k0+16+8g+(j-8) ----
        if (coutA < Cout)
            __builtin_prefetch(rowA + k0 + 32, 0, 1);   // next weight chunk
        v16h aF;
        #pragma unroll
        for (int j = 0; j < 8; ++j) {
            int kk = k0 + 8 * g + j;
            float w = (coutA < Cout && kk < Ktot) ? rowA[kk] : 0.f;
            aF[j] = (_Float16)w;
        }
        #pragma unroll
        for (int j = 0; j < 8; ++j) {
            int kk = k0 + 16 + 8 * g + j;
            float w = (coutA < Cout && kk < Ktot) ? rowA[kk] : 0.f;
            aF[8 + j] = (_Float16)w;
        }

        // ---- B fragment from LDS: j=0..7 -> K=8g+j ; j=8..15 -> K=16+8g+(j-8) at N = hl ----
        v16h bF;
        #pragma unroll
        for (int j = 0; j < 16; ++j) bF[j] = sB[hl][8 * g + j + (j >= 8 ? 8 : 0)];

        acc = __builtin_amdgcn_wmma_f32_16x16x32_f16(
            /*neg_a=*/false, aF, /*neg_b=*/false, bF,
            /*c_mod=*/(short)0, acc, /*reuse_a=*/false, /*reuse_b=*/false);
    }

    // ---- store D: acc[r] -> M = tileM + r + 8*g, N pixel = tileN + hl ----
    #pragma unroll
    for (int r = 0; r < 8; ++r) {
        int m = tileM + r + 8 * g;
        if (m >= Cout) continue;
        float val = acc[r] + bias[m];
        if (act == 1) {
            float a = alpha[0];
            val = (val > 0.f) ? val : a * val;
        } else if (act == 2) {
            val = 2.0f * tanhf(val);
        } else if (act == 3) {
            float x = val;
            val = 0.5f * x * (1.0f + tanhf(0.7978845608028654f *
                                           (x + 0.044715f * x * x * x)));
        }
        size_t oidx = ((size_t)(bimg * Cout + m)) * HW + rem;
        if (res) val += res[oidx];
        out[oidx] = val;
    }
}

// ---------------------------------------------------------------------------
// x = concat([feat_t, bwarp(fx, ftx), ftx], channel dim) -> (B, 2C+2, H, W)
// ---------------------------------------------------------------------------
__global__ void bwarp_concat_kernel(const float* __restrict__ feat_t,
                                    const float* __restrict__ fx,
                                    const float* __restrict__ ftx,
                                    float* __restrict__ xcat,
                                    int Bn, int Hh, int Ww)
{
    const int C = 144, HW = Hh * Ww, CC = 2 * C + 2;
    size_t idx = (size_t)blockIdx.x * blockDim.x + threadIdx.x;
    size_t total = (size_t)Bn * CC * HW;
    if (idx >= total) return;
    int d  = (int)(idx % HW);
    int ch = (int)((idx / HW) % CC);
    int b  = (int)(idx / ((size_t)HW * CC));
    int pxi = d % Ww, pyi = d / Ww;
    float val;
    if (ch < C) {
        val = feat_t[((size_t)(b * C + ch)) * HW + d];
    } else if (ch < 2 * C) {
        float fxv = ftx[((size_t)(b * 2 + 0)) * HW + d];
        float fyv = ftx[((size_t)(b * 2 + 1)) * HW + d];
        val = bilin_zero(fx + ((size_t)(b * C + (ch - C))) * HW, Hh, Ww,
                         (float)pxi + fxv, (float)pyi + fyv);
    } else {
        val = ftx[((size_t)(b * 2 + (ch - 2 * C))) * HW + d];
    }
    xcat[idx] = val;
}

// ---------------------------------------------------------------------------
// Fused deformable sampling + folded k/v projections + softmax, per pixel.
//   logit[h,s] = (sum_{c in h} q_c * Wk[c,:]) . samp_s + q_h . bk_h
//   at[c]      = Wv[c,:] . (sum_s attn[h(c),s] samp_s) + bv[c]
// ---------------------------------------------------------------------------
__global__ __launch_bounds__(256)
void attn_fused_kernel(const float* __restrict__ qbuf,
                       const float* __restrict__ feat0, const float* __restrict__ feat1,
                       const float* __restrict__ off0,  const float* __restrict__ off1,
                       const float* __restrict__ ft0,   const float* __restrict__ ft1,
                       const float* __restrict__ kw,    const float* __restrict__ kb,
                       const float* __restrict__ vw,    const float* __restrict__ vb,
                       float* __restrict__ atbuf, int Bn, int Hh, int Ww)
{
    const int HW = Hh * Ww;
    const int pix = blockIdx.x;
    const int b = pix / HW;
    const int d = pix % HW;
    const int pxi = d % Ww, pyi = d / Ww;
    const int tid = threadIdx.x;

    __shared__ float s_samp[144 * 18];   // [cin*18 + s]
    __shared__ float s_q[144];
    __shared__ float s_u[12 * 144];      // folded q.Wk, [h*144 + cin]
    __shared__ float s_qb[12];
    __shared__ float s_logit[12 * 18];
    __shared__ float s_wsum[12 * 144];

    // 1. deformable bilinear sampling, 18 samples x 144 channels
    for (int i = tid; i < 144 * 18; i += 256) {
        int c = i / 18, s = i % 18;
        const float* feat = (s < 9) ? feat0 : feat1;
        const float* offp = (s < 9) ? off0  : off1;
        const float* ftp  = (s < 9) ? ft0   : ft1;
        int sl = s % 9;
        int gg = c / 12;                       // sampling group
        int ch2 = (gg * 9 + sl) * 2;
        float flow_x = offp[((size_t)(b * 216 + ch2    )) * HW + d] +
                       ftp[((size_t)(b * 2 + 0)) * HW + d];
        float flow_y = offp[((size_t)(b * 216 + ch2 + 1)) * HW + d] +
                       ftp[((size_t)(b * 2 + 1)) * HW + d];
        float base_x = -1.f + 2.f * (float)pxi / (float)(Ww - 1);
        float base_y = -1.f + 2.f * (float)pyi / (float)(Hh - 1);
        // reference's asymmetric normalization, then un-normalize (align_corners)
        float gridx = base_x + flow_x / ((float)(Ww - 1) * 2.f);
        float gridy = base_y + flow_y * 2.f / (float)(Hh - 1);
        float gx = (gridx + 1.f) * 0.5f * (float)(Ww - 1);
        float gy = (gridy + 1.f) * 0.5f * (float)(Hh - 1);
        s_samp[i] = bilin_zero(feat + ((size_t)(b * 144 + c)) * HW, Hh, Ww, gx, gy);
    }
    if (tid < 144) s_q[tid] = qbuf[((size_t)(b * 144 + tid)) * HW + d];
    __syncthreads();

    // 2. fold q into Wk
    for (int i = tid; i < 12 * 144; i += 256) {
        int h = i / 144, cin = i % 144;
        float acc = 0.f;
        #pragma unroll
        for (int j = 0; j < 12; ++j)
            acc += s_q[h * 12 + j] * kw[(size_t)(h * 12 + j) * 144 + cin];
        s_u[i] = acc;
    }
    if (tid < 12) {
        float acc = 0.f;
        #pragma unroll
        for (int j = 0; j < 12; ++j) acc += s_q[tid * 12 + j] * kb[tid * 12 + j];
        s_qb[tid] = acc;
    }
    __syncthreads();

    // 3. logits
    const float scale = 0.28867513459481287f; // (144/12)^-0.5
    for (int i = tid; i < 12 * 18; i += 256) {
        int h = i / 18, s = i % 18;
        float acc = s_qb[h];
        const float* u = &s_u[h * 144];
        for (int cin = 0; cin < 144; ++cin) acc += u[cin] * s_samp[cin * 18 + s];
        s_logit[i] = acc * scale;
    }
    __syncthreads();

    // 4. softmax over s per head
    if (tid < 12) {
        float mx = -1e30f;
        for (int s = 0; s < 18; ++s) mx = fmaxf(mx, s_logit[tid * 18 + s]);
        float sum = 0.f;
        for (int s = 0; s < 18; ++s) {
            float e = expf(s_logit[tid * 18 + s] - mx);
            s_logit[tid * 18 + s] = e; sum += e;
        }
        float inv = 1.f / sum;
        for (int s = 0; s < 18; ++s) s_logit[tid * 18 + s] *= inv;
    }
    __syncthreads();

    // 5. attn-weighted sample sum per head
    for (int i = tid; i < 12 * 144; i += 256) {
        int h = i / 144, cin = i % 144;
        float acc = 0.f;
        #pragma unroll
        for (int s = 0; s < 18; ++s)
            acc += s_logit[h * 18 + s] * s_samp[cin * 18 + s];
        s_wsum[i] = acc;
    }
    __syncthreads();

    // 6. v projection
    for (int c = tid; c < 144; c += 256) {
        int h = c / 12;
        float acc = vb[c];
        const float* wsum = &s_wsum[h * 144];
        for (int cin = 0; cin < 144; ++cin) acc += vw[(size_t)c * 144 + cin] * wsum[cin];
        atbuf[((size_t)(b * 144 + c)) * HW + d] = acc;
    }
}

// ---------------------------------------------------------------------------
// ConvTranspose2d(144->2, k=4, s=2, p=1)(mov) + 2 * bilinear_resize(ftx, x2)
// ---------------------------------------------------------------------------
__global__ void upflow_kernel(const float* __restrict__ mov, const float* __restrict__ ftx,
                              const float* __restrict__ wflow, const float* __restrict__ bflow,
                              float* __restrict__ outp, int Bn, int Hh, int Ww)
{
    const int OH = 2 * Hh, OW = 2 * Ww;
    size_t idx = (size_t)blockIdx.x * blockDim.x + threadIdx.x;
    size_t total = (size_t)Bn * 2 * OH * OW;
    if (idx >= total) return;
    int ox = (int)(idx % OW);
    int oy = (int)((idx / OW) % OH);
    int ch = (int)((idx / ((size_t)OW * OH)) % 2);
    int b  = (int)(idx / ((size_t)OW * OH * 2));

    // transpose conv: oy = 2*iy + ky - 1
    float acc = bflow[ch];
    #pragma unroll
    for (int ky = 0; ky < 4; ++ky) {
        int ty = oy + 1 - ky;
        if (ty < 0 || (ty & 1)) continue;
        int iy = ty >> 1;
        if (iy >= Hh) continue;
        #pragma unroll
        for (int kx = 0; kx < 4; ++kx) {
            int tx = ox + 1 - kx;
            if (tx < 0 || (tx & 1)) continue;
            int ix = tx >> 1;
            if (ix >= Ww) continue;
            const float* mcol = mov + ((size_t)b * 144) * Hh * Ww + (size_t)iy * Ww + ix;
            for (int ic = 0; ic < 144; ++ic)
                acc += mcol[(size_t)ic * Hh * Ww] * wflow[((ic * 2 + ch) * 4 + ky) * 4 + kx];
        }
    }

    // bilinear x2 upsample of ftx (half-pixel centers, edge clamp)
    float sx = ((float)ox + 0.5f) * 0.5f - 0.5f;
    float sy = ((float)oy + 0.5f) * 0.5f - 0.5f;
    float x0f = floorf(sx), y0f = floorf(sy);
    float wx = sx - x0f, wy = sy - y0f;
    int x0 = (int)x0f, y0 = (int)y0f;
    int x0c = x0 < 0 ? 0 : (x0 > Ww - 1 ? Ww - 1 : x0);
    int x1c = x0 + 1 < 0 ? 0 : (x0 + 1 > Ww - 1 ? Ww - 1 : x0 + 1);
    int y0c = y0 < 0 ? 0 : (y0 > Hh - 1 ? Hh - 1 : y0);
    int y1c = y0 + 1 < 0 ? 0 : (y0 + 1 > Hh - 1 ? Hh - 1 : y0 + 1);
    const float* f = ftx + ((size_t)(b * 2 + ch)) * Hh * Ww;
    float up = (1.f - wx) * (1.f - wy) * f[y0c * Ww + x0c] +
               wx * (1.f - wy) * f[y0c * Ww + x1c] +
               (1.f - wx) * wy * f[y1c * Ww + x0c] +
               wx * wy * f[y1c * Ww + x1c];

    outp[idx] = acc + 2.f * up;
}

// ---------------------------------------------------------------------------
extern "C" void kernel_launch(void* const* d_in, const int* in_sizes, int n_in,
                              void* d_out, int out_size, void* d_ws, size_t ws_size,
                              hipStream_t stream) {
    (void)in_sizes; (void)n_in; (void)out_size; (void)ws_size;
    const float* me0_w  = (const float*)d_in[0];
    const float* me0_b  = (const float*)d_in[1];
    const float* me0_a  = (const float*)d_in[2];
    const float* me1_w  = (const float*)d_in[3];
    const float* me1_b  = (const float*)d_in[4];
    const float* me1_a  = (const float*)d_in[5];
    const float* rb1_w  = (const float*)d_in[6];
    const float* rb1_b  = (const float*)d_in[7];
    const float* rb_a   = (const float*)d_in[8];
    const float* rb2_w  = (const float*)d_in[9];
    const float* rb2_b  = (const float*)d_in[10];
    const float* off_w  = (const float*)d_in[11];
    const float* off_b  = (const float*)d_in[12];
    const float* flow_w = (const float*)d_in[13];
    const float* flow_b = (const float*)d_in[14];
    const float* q_w    = (const float*)d_in[15];
    const float* q_b    = (const float*)d_in[16];
    const float* k_w    = (const float*)d_in[17];
    const float* k_b    = (const float*)d_in[18];
    const float* v_w    = (const float*)d_in[19];
    const float* v_b    = (const float*)d_in[20];
    const float* f1_w   = (const float*)d_in[21];
    const float* f1_b   = (const float*)d_in[22];
    const float* f2_w   = (const float*)d_in[23];
    const float* f2_b   = (const float*)d_in[24];
    const float* feat_t = (const float*)d_in[25];
    const float* feat0  = (const float*)d_in[26];
    const float* feat1  = (const float*)d_in[27];
    const float* ft0    = (const float*)d_in[28];
    const float* ft1    = (const float*)d_in[29];

    const int Bn = 2, C = 144, Hh = 96, Ww = 96;
    const int HW = Hh * Ww;

    float* ws = (float*)d_ws;
    size_t o = 0;
    auto alloc = [&](size_t n) { float* p = ws + o; o += n; return p; };
    float* xcat = alloc((size_t)Bn * (2 * C + 2) * HW);
    float* t1   = alloc((size_t)Bn * 288 * HW);
    float* x2   = alloc((size_t)Bn * 144 * HW);
    float* hbuf = alloc((size_t)Bn * 72 * HW);
    float* mov0 = alloc((size_t)Bn * 144 * HW);
    float* mov1 = alloc((size_t)Bn * 144 * HW);
    float* off0 = alloc((size_t)Bn * 216 * HW);
    float* off1 = alloc((size_t)Bn * 216 * HW);
    float* qbuf = alloc((size_t)Bn * 144 * HW);
    float* atb  = alloc((size_t)Bn * 144 * HW);
    float* mlp1 = alloc((size_t)Bn * 288 * HW);

    float* out_main = (float*)d_out;                            // (B,144,96,96)
    float* flowo0 = out_main + (size_t)Bn * 144 * HW;           // (B,2,192,192)
    float* flowo1 = flowo0 + (size_t)Bn * 2 * (2 * Hh) * (2 * Ww);

    dim3 cblk(32 * CONV_WAVES);
    auto conv = [&](const float* in, const float* w, const float* b, const float* res,
                    const float* alpha, float* outp, int Cin, int Cout, int Kk, int pad,
                    int act) {
        dim3 grid((unsigned)((Cout + 16 * CONV_WAVES - 1) / (16 * CONV_WAVES)),
                  (unsigned)((Bn * HW) / 16));
        conv_wmma_kernel<<<grid, cblk, 0, stream>>>(in, w, b, res, alpha, outp,
                                                    Bn, Cin, Cout, Hh, Ww, Kk, pad, act);
    };

    // movement() x2 (buffers xcat/t1/x2/hbuf reused per path)
    for (int path = 0; path < 2; ++path) {
        const float* fx  = path ? feat1 : feat0;
        const float* fl  = path ? ft1 : ft0;
        float* mov = path ? mov1 : mov0;
        float* off = path ? off1 : off0;

        size_t tot = (size_t)Bn * (2 * C + 2) * HW;
        bwarp_concat_kernel<<<(unsigned)((tot + 255) / 256), 256, 0, stream>>>(
            feat_t, fx, fl, xcat, Bn, Hh, Ww);

        conv(xcat, me0_w, me0_b, nullptr, me0_a, t1,   290, 288, 3, 1, 1); // prelu
        conv(t1,   me1_w, me1_b, nullptr, me1_a, x2,   288, 144, 3, 1, 1); // prelu
        conv(x2,   rb1_w, rb1_b, nullptr, rb_a,  hbuf, 144,  72, 5, 2, 1); // prelu
        conv(hbuf, rb2_w, rb2_b, x2,     nullptr, mov,  72, 144, 5, 2, 0); // +res
        conv(mov,  off_w, off_b, nullptr, nullptr, off, 144, 216, 3, 1, 2); // 2*tanh
    }

    // q projection (WMMA 1x1)
    conv(feat_t, q_w, q_b, nullptr, nullptr, qbuf, 144, 144, 1, 0, 0);

    // fused deformable sampling + attention
    attn_fused_kernel<<<(unsigned)(Bn * HW), 256, 0, stream>>>(
        qbuf, feat0, feat1, off0, off1, ft0, ft1, k_w, k_b, v_w, v_b, atb, Bn, Hh, Ww);

    // MLP (WMMA 1x1): out = at + f2(gelu(f1(at)))
    conv(atb,  f1_w, f1_b, nullptr, nullptr, mlp1,     144, 288, 1, 0, 3); // gelu
    conv(mlp1, f2_w, f2_b, atb,    nullptr, out_main,  288, 144, 1, 0, 0); // +res

    // flow upsampling
    {
        size_t tot = (size_t)Bn * 2 * (2 * Hh) * (2 * Ww);
        unsigned gsz = (unsigned)((tot + 255) / 256);
        upflow_kernel<<<gsz, 256, 0, stream>>>(mov0, ft0, flow_w, flow_b, flowo0, Bn, Hh, Ww);
        upflow_kernel<<<gsz, 256, 0, stream>>>(mov1, ft1, flow_w, flow_b, flowo1, Bn, Hh, Ww);
    }
}
